// SparseDeformableTransformer_7318624272914
// MI455X (gfx1250) — compile-verified, matrix operand order checked by reading, not run on hardware
//
#include <hip/hip_runtime.h>

// ---------------- problem constants ----------------
constexpr int kBS = 32, kNQ = 300, kD = 256, kH = 8, kHD = 32, kP = 4;
constexpr int kLV = 4, kS = 3840, kNL = 6, kDFF = 1024;
constexpr int kQKV = 3 * kD;                 // 768
constexpr int kRows = kBS * kNQ;             // 9600
constexpr int kNQP = 320;                    // padded attention row length (10 * 32)
constexpr long kInterOff = (long)kNL * kRows * kD;          // 14,745,600
constexpr long kRefOff   = kInterOff;
constexpr long kLocOff   = kRefOff + (long)kNL * kRows;      // 14,803,200
constexpr long kAwOff    = kLocOff + (long)kBS * kNL * kNQ * kH * kLV * kP;

typedef __attribute__((ext_vector_type(16))) __bf16 v16bf;
typedef __attribute__((ext_vector_type(8)))  float  v8f;
typedef __attribute__((ext_vector_type(4)))  float  v4f;
typedef __attribute__((ext_vector_type(4)))  __bf16 v4bf;

union FragB16 {
  v16bf v;
  unsigned short u[16];
  uint4 q[2];
};

// hardware bf16 conversion (v_cvt_pk_bf16_f32 class), not software bit tricks
__device__ __forceinline__ unsigned short f2bfu(float f) {
  union { __bf16 b; unsigned short s; } z;
  z.b = (__bf16)f;
  return z.s;
}
__device__ __forceinline__ uint2 pk4(float4 f) {
  v4f x = {f.x, f.y, f.z, f.w};
  v4bf y = __builtin_convertvector(x, v4bf);
  union { v4bf v; uint2 u; } z;
  z.v = y;
  return z.u;
}
__device__ __forceinline__ uint4 pk8(float4 a, float4 b) {
  uint2 u = pk4(a), v = pk4(b);
  return make_uint4(u.x, u.y, v.x, v.y);
}

__device__ __forceinline__ v16bf frag_from_lds(const unsigned short* base, int half) {
  FragB16 f;
  f.q[0] = *(const uint4*)(base + half * 8);        // K = half*8 .. +7
  f.q[1] = *(const uint4*)(base + 16 + half * 8);   // K = 16+half*8 .. +7
  return f.v;
}
__device__ __forceinline__ v16bf frag_from_global(const float* row, int half) {
  // row points at 16-byte-aligned fp32 data; build bf16 frag for K=0..31
  float4 a0 = *(const float4*)(row + half * 8);
  float4 a1 = *(const float4*)(row + half * 8 + 4);
  float4 a2 = *(const float4*)(row + 16 + half * 8);
  float4 a3 = *(const float4*)(row + 16 + half * 8 + 4);
  FragB16 f;
  f.q[0] = pk8(a0, a1);
  f.q[1] = pk8(a2, a3);
  return f.v;
}
__device__ __forceinline__ v8f wmma_bf16(v16bf a, v16bf b, v8f c) {
  return __builtin_amdgcn_wmma_f32_16x16x32_bf16(false, a, false, b, (short)0, c,
                                                 false, false);
}

// ---------------- dense GEMM: C[M,N] = A[M,K] @ B[N,K]^T + bias ----------------
// REQUIRES: M % 128 == 0, N % 64 == 0, K % 32 == 0 (true for all call sites).
// block tile 128x64, 8 waves (4M x 2N), each wave 32x32 => 4 WMMA per K-step.
__global__ __launch_bounds__(256) void gemm_btn_kernel(
    const float* __restrict__ A, const float* __restrict__ B,
    const float* __restrict__ bias, float* __restrict__ C,
    int M, int N, int K, int ldc, int relu) {
  __shared__ unsigned short sa[128 * 32];
  __shared__ unsigned short sb[64 * 32];
  const int tid  = threadIdx.x;
  const int lane = tid & 31;
  const int w    = tid >> 5;
  const int wm   = w >> 1;     // 0..3 -> 32-row slab
  const int wn   = w & 1;      // 0..1 -> 32-col slab
  const int mBase = blockIdx.y * 128;
  const int nBase = blockIdx.x * 64;
  const int half = lane >> 4, l16 = lane & 15;

  // A loader: thread -> (row 0..127, 16 cols), B loader: thread -> (row 0..63, 8 cols)
  const int ar = tid >> 1, ac = (tid & 1) * 16;
  const int br = tid >> 2, bc = (tid & 3) * 8;
  const float* apRow = A + (long)(mBase + ar) * K + ac;
  const float* bpRow = B + (long)(nBase + br) * K + bc;
  unsigned short* sap = sa + ar * 32 + ac;
  unsigned short* sbp = sb + br * 32 + bc;

  v8f c00 = {}, c01 = {}, c10 = {}, c11 = {};

  for (int kt = 0; kt < K; kt += 32) {
    {
      const float4* ap = (const float4*)(apRow + kt);
      float4 f0 = ap[0], f1 = ap[1], f2 = ap[2], f3 = ap[3];
      *(uint4*)(sap)     = pk8(f0, f1);
      *(uint4*)(sap + 8) = pk8(f2, f3);
      const float4* bp = (const float4*)(bpRow + kt);
      float4 g0 = bp[0], g1 = bp[1];
      *(uint4*)(sbp) = pk8(g0, g1);
      if (kt + 32 < K) {
        __builtin_prefetch(apRow + kt + 32, 0, 0);
        __builtin_prefetch(bpRow + kt + 32, 0, 0);
      }
    }
    __syncthreads();

    v16bf a0 = frag_from_lds(sa + (wm * 32 + l16) * 32, half);
    v16bf a1 = frag_from_lds(sa + (wm * 32 + 16 + l16) * 32, half);
    v16bf b0 = frag_from_lds(sb + (wn * 32 + l16) * 32, half);
    v16bf b1 = frag_from_lds(sb + (wn * 32 + 16 + l16) * 32, half);
    c00 = wmma_bf16(a0, b0, c00);
    c01 = wmma_bf16(a0, b1, c01);
    c10 = wmma_bf16(a1, b0, c10);
    c11 = wmma_bf16(a1, b1, c11);
    __syncthreads();
  }

  const int n0 = nBase + wn * 32 + l16;
  const int n1 = n0 + 16;
  const float bv0 = bias ? bias[n0] : 0.f;
  const float bv1 = bias ? bias[n1] : 0.f;
#pragma unroll
  for (int r = 0; r < 8; ++r) {
    const int m0 = mBase + wm * 32 + r + half * 8;
    const int m1 = m0 + 16;
    float v00 = c00[r] + bv0, v01 = c01[r] + bv1;
    float v10 = c10[r] + bv0, v11 = c11[r] + bv1;
    if (relu) {
      v00 = fmaxf(v00, 0.f); v01 = fmaxf(v01, 0.f);
      v10 = fmaxf(v10, 0.f); v11 = fmaxf(v11, 0.f);
    }
    C[(long)m0 * ldc + n0] = v00;
    C[(long)m0 * ldc + n1] = v01;
    C[(long)m1 * ldc + n0] = v10;
    C[(long)m1 * ldc + n1] = v11;
  }
}

// ---------------- attention scores: sc[b,h,q,k] = qh . kh / sqrt(HD) ------------
// one wave computes a 16x16 score tile; K = HD = 32 = exactly one WMMA step.
__global__ __launch_bounds__(32) void attn_scores_kernel(
    const float* __restrict__ qkv, float* __restrict__ attn) {
  const int bh = blockIdx.z;           // 0..255
  const int b = bh >> 3, h = bh & 7;
  const int qt = blockIdx.y * 16, kt = blockIdx.x * 16;
  const int lane = threadIdx.x & 31;
  const int half = lane >> 4, l16 = lane & 15;

  const int qr = qt + l16 < kNQ ? qt + l16 : kNQ - 1;
  const int kr = kt + l16 < kNQ ? kt + l16 : kNQ - 1;
  const float* qrow = qkv + (long)(b * kNQ + qr) * kQKV + h * kHD;          // Q
  const float* krow = qkv + (long)(b * kNQ + kr) * kQKV + kD + h * kHD;     // K

  v16bf fa = frag_from_global(qrow, half);
  v16bf fb = frag_from_global(krow, half);
  v8f c = {};
  c = wmma_bf16(fa, fb, c);

  const float scale = 0.17677669529663687f;  // 1/sqrt(32)
  float* out = attn + (long)bh * kNQ * kNQP;
  const int n = kt + l16;
#pragma unroll
  for (int r = 0; r < 8; ++r) {
    const int m = qt + r + half * 8;
    if (m < kNQ && n < kNQ) out[(long)m * kNQP + n] = c[r] * scale;
  }
}

// ---------------- masked softmax over rows of NQ (stride NQP, zero the pad) -----
__global__ __launch_bounds__(256) void softmax_mask_kernel(
    float* __restrict__ attn, const unsigned char* __restrict__ mask) {
  const int lane = threadIdx.x & 31;
  const int row = blockIdx.x * 8 + (threadIdx.x >> 5);   // 0..76799
  const int b = row / (kH * kNQ);
  float* p = attn + (long)row * kNQP;
  const unsigned char* mb = mask + b * kNQ;

  float mx = -1e30f;
  for (int k = lane; k < kNQ; k += 32) {
    float s = mb[k] ? p[k] : -1e9f;
    mx = fmaxf(mx, s);
  }
  for (int o = 16; o; o >>= 1) mx = fmaxf(mx, __shfl_xor(mx, o, 32));

  float sum = 0.f;
  for (int k = lane; k < kNQ; k += 32) {
    float s = mb[k] ? p[k] : -1e9f;
    float e = __expf(s - mx);
    p[k] = e;
    sum += e;
  }
  for (int o = 16; o; o >>= 1) sum += __shfl_xor(sum, o, 32);
  const float inv = 1.f / sum;
  for (int k = lane; k < kNQ; k += 32) p[k] *= inv;
  for (int k = kNQ + lane; k < kNQP; k += 32) p[k] = 0.f;   // pad = 0 prob
}

// ---------------- attention apply: t2[b,q,h*32+d] = sum_k at[q,k] v[k,d] --------
// guard-free K loop over NQP; A pad is exactly zero, B rows clamped.
__global__ __launch_bounds__(256) void attn_apply_kernel(
    const float* __restrict__ attn, const float* __restrict__ qkv,
    float* __restrict__ t2) {
  __shared__ unsigned short sbt[32 * 32];  // [d][k] transposed V tile
  const int bh = blockIdx.y;
  const int b = bh >> 3, h = bh & 7;
  const int tid = threadIdx.x, lane = tid & 31, w = tid >> 5;
  const int half = lane >> 4, l16 = lane & 15;
  const int qRow = blockIdx.x * 128 + w * 16;
  const float* at = attn + (long)bh * kNQ * kNQP;
  const int q = qRow + l16 < kNQ ? qRow + l16 : kNQ - 1;
  const float* aRowBase = at + (long)q * kNQP;

  // V-tile loader: thread -> (k 0..31, 4 consecutive d)
  const int vk = tid >> 3, vd = (tid & 7) * 4;

  v8f c0 = {}, c1 = {};
  for (int kt = 0; kt < kNQP; kt += 32) {
    {
      const int gk = kt + vk < kNQ ? kt + vk : kNQ - 1;  // clamp (A pad is 0)
      float4 v = *(const float4*)(qkv + (long)(b * kNQ + gk) * kQKV + 2 * kD +
                                  h * kHD + vd);
      sbt[(vd + 0) * 32 + vk] = f2bfu(v.x);
      sbt[(vd + 1) * 32 + vk] = f2bfu(v.y);
      sbt[(vd + 2) * 32 + vk] = f2bfu(v.z);
      sbt[(vd + 3) * 32 + vk] = f2bfu(v.w);
    }
    __syncthreads();

    v16bf fa = frag_from_global(aRowBase + kt, half);
    v16bf b0 = frag_from_lds(sbt + l16 * 32, half);
    v16bf b1 = frag_from_lds(sbt + (16 + l16) * 32, half);
    c0 = wmma_bf16(fa, b0, c0);
    c1 = wmma_bf16(fa, b1, c1);
    __syncthreads();
  }

#pragma unroll
  for (int r = 0; r < 8; ++r) {
    const int m = qRow + r + half * 8;
    if (m < kNQ) {
      t2[(long)(b * kNQ + m) * kD + h * kHD + l16]      = c0[r];
      t2[(long)(b * kNQ + m) * kD + h * kHD + 16 + l16] = c1[r];
    }
  }
}

// ---------------- layernorm of residual sum: y = LN(a+b)*w + bias --------------
__global__ __launch_bounds__(256) void ln_res_kernel(
    const float* __restrict__ a, const float* __restrict__ b,
    const float* __restrict__ w, const float* __restrict__ bias,
    float* __restrict__ y, int rows) {
  const int lane = threadIdx.x & 31;
  const int row = blockIdx.x * 8 + (threadIdx.x >> 5);
  if (row >= rows) return;
  const float* pa = a + (long)row * kD;
  const float* pb = b + (long)row * kD;
  float v[8];
  float s = 0.f;
#pragma unroll
  for (int i = 0; i < 8; ++i) {
    const int c = lane + 32 * i;
    v[i] = pa[c] + pb[c];
    s += v[i];
  }
  for (int o = 16; o; o >>= 1) s += __shfl_xor(s, o, 32);
  const float mu = s * (1.f / kD);
  float qv = 0.f;
#pragma unroll
  for (int i = 0; i < 8; ++i) {
    const float d = v[i] - mu;
    qv += d * d;
  }
  for (int o = 16; o; o >>= 1) qv += __shfl_xor(qv, o, 32);
  const float rstd = rsqrtf(qv * (1.f / kD) + 1e-5f);
#pragma unroll
  for (int i = 0; i < 8; ++i) {
    const int c = lane + 32 * i;
    y[(long)row * kD + c] = (v[i] - mu) * rstd * w[c] + bias[c];
  }
}

// ---------------- elementwise helpers ----------------
__global__ void add_pos_kernel(const float* __restrict__ x,
                               const float* __restrict__ pos,
                               float* __restrict__ y, long total, int posmod) {
  const long i = (long)blockIdx.x * 256 + threadIdx.x;
  if (i < total) y[i] = x[i] + pos[i % posmod];
}

__global__ void mask_value_kernel(float* __restrict__ value,
                                  const unsigned char* __restrict__ pad) {
  const long r = blockIdx.x;
  const float m = pad[r] ? 0.f : 1.f;
  value[r * kD + threadIdx.x] *= m;
}

__global__ void copy_kernel(const float* __restrict__ src,
                            float* __restrict__ dst, long n4) {
  const long i = (long)blockIdx.x * 256 + threadIdx.x;
  if (i < n4) ((float4*)dst)[i] = ((const float4*)src)[i];
}

__global__ void ref_out_kernel(const float* __restrict__ refp,
                               float* __restrict__ dst) {
  const int i = blockIdx.x * 256 + threadIdx.x;
  if (i < kRows) {
    const float v = refp[i];
#pragma unroll
    for (int l = 0; l < kNL; ++l) dst[(long)l * kRows + i] = v;
  }
}

// ---------------- deformable sampling: one wave per (b,q,h), lane = channel ----
__global__ __launch_bounds__(256) void msdeform_kernel(
    const float* __restrict__ boff, const float* __restrict__ baw,
    const float* __restrict__ value, const float* __restrict__ refp,
    const float* __restrict__ svr, const int* __restrict__ tshapes,
    float* __restrict__ bms, float* __restrict__ out_locs,
    float* __restrict__ out_aws, int layer) {
  const int g = blockIdx.x * 8 + (threadIdx.x >> 5);
  if (g >= kBS * kNQ * kH) return;
  const int h = g & 7;
  const int q = (g >> 3) % kNQ;
  const int b = g / (kNQ * kH);
  const int lane = threadIdx.x & 31;

  const long rowBase = (long)(b * kNQ + q) * (kH * kLV * kP);
  const float* awr = baw + rowBase + h * (kLV * kP);
  const float* ofr = boff + rowBase + h * (kLV * kP);

  float mx = -1e30f;
#pragma unroll
  for (int j = 0; j < 16; ++j) mx = fmaxf(mx, awr[j]);
  float e[16];
  float se = 0.f;
#pragma unroll
  for (int j = 0; j < 16; ++j) {
    e[j] = __expf(awr[j] - mx);
    se += e[j];
  }
  const float inv = 1.f / se;
  const float rp = refp[b * kNQ + q];

  float acc = 0.f;
  int start = 0;
#pragma unroll
  for (int lvl = 0; lvl < kLV; ++lvl) {
    const int L = tshapes[lvl];
    const float fL = (float)L;
    const float ri = rp * svr[b * kLV + lvl];
#pragma unroll
    for (int p = 0; p < kP; ++p) {
      const int j = lvl * kP + p;
      const float loc = ri + ofr[j] / fL;
      const float aw = e[j] * inv;
      if (lane == 0) {
        const long oi =
            ((((long)b * kNL + layer) * kNQ + q) * kH + h) * (kLV * kP) + j;
        out_locs[oi] = loc;
        out_aws[oi] = aw;
      }
      const float xf = loc * fL - 0.5f;
      const float x0 = floorf(xf);
      const float wgt = xf - x0;
      const int i0 = (int)x0;
      // branch-free: clamp index, multiply by in-range mask
      const int c0 = min(max(i0, 0), L - 1);
      const int c1 = min(max(i0 + 1, 0), L - 1);
      const float m0 = (i0 >= 0 && i0 <= L - 1) ? 1.f : 0.f;
      const float m1 = (i0 + 1 >= 0 && i0 + 1 <= L - 1) ? 1.f : 0.f;
      const float v0 = value[(long)(b * kS + start + c0) * kD + h * kHD + lane] * m0;
      const float v1 = value[(long)(b * kS + start + c1) * kD + h * kHD + lane] * m1;
      acc += aw * ((1.f - wgt) * v0 + wgt * v1);
    }
    start += L;
  }
  bms[(long)(b * kNQ + q) * kD + h * kHD + lane] = acc;
}

// ---------------- host orchestration ----------------
static inline void gemm(const float* A, const float* B, const float* bias,
                        float* C, int M, int N, int K, int ldc, int relu,
                        hipStream_t s) {
  dim3 grid(N / 64, M / 128);
  gemm_btn_kernel<<<grid, 256, 0, s>>>(A, B, bias, C, M, N, K, ldc, relu);
}

extern "C" void kernel_launch(void* const* d_in, const int* in_sizes, int n_in,
                              void* d_out, int out_size, void* d_ws,
                              size_t ws_size, hipStream_t stream) {
  (void)in_sizes; (void)n_in; (void)out_size; (void)ws_size;
  const float* tgt    = (const float*)d_in[0];
  const float* refp   = (const float*)d_in[1];
  const float* src    = (const float*)d_in[2];
  const float* svr    = (const float*)d_in[3];
  const float* qpos   = (const float*)d_in[4];
  const int*   tshape = (const int*)d_in[5];
  const unsigned char* src_pad = (const unsigned char*)d_in[7];
  const unsigned char* q_mask  = (const unsigned char*)d_in[8];

  const float* p_sa_w_in  = (const float*)d_in[9];
  const float* p_sa_b_in  = (const float*)d_in[10];
  const float* p_sa_w_out = (const float*)d_in[11];
  const float* p_sa_b_out = (const float*)d_in[12];
  const float* p_off_w = (const float*)d_in[13];
  const float* p_off_b = (const float*)d_in[14];
  const float* p_aw_w  = (const float*)d_in[15];
  const float* p_aw_b  = (const float*)d_in[16];
  const float* p_val_w = (const float*)d_in[17];
  const float* p_val_b = (const float*)d_in[18];
  const float* p_mso_w = (const float*)d_in[19];
  const float* p_mso_b = (const float*)d_in[20];
  const float* p_ff1_w = (const float*)d_in[21];
  const float* p_ff1_b = (const float*)d_in[22];
  const float* p_ff2_w = (const float*)d_in[23];
  const float* p_ff2_b = (const float*)d_in[24];
  const float* p_n1_w = (const float*)d_in[25];
  const float* p_n1_b = (const float*)d_in[26];
  const float* p_n2_w = (const float*)d_in[27];
  const float* p_n2_b = (const float*)d_in[28];
  const float* p_n3_w = (const float*)d_in[29];
  const float* p_n3_b = (const float*)d_in[30];

  float* out = (float*)d_out;
  float* ws = (float*)d_ws;
  size_t off = 0;
  auto alloc = [&](size_t n) { float* p = ws + off; off += n; return p; };
  float* bx0   = alloc((size_t)kRows * kD);
  float* bx1   = alloc((size_t)kRows * kD);
  float* bq    = alloc((size_t)kRows * kD);
  float* bqkv  = alloc((size_t)kRows * kQKV);
  float* battn = alloc((size_t)kBS * kH * kNQ * kNQP);
  float* bt2   = alloc((size_t)kRows * kD);
  float* bxA   = alloc((size_t)kRows * kD);
  float* bval  = alloc((size_t)kBS * kS * kD);
  float* boff  = alloc((size_t)kRows * kH * kLV * kP);
  float* baw   = alloc((size_t)kRows * kH * kLV * kP);
  float* bms   = alloc((size_t)kRows * kD);
  float* bff   = alloc((size_t)kRows * kDFF);

  const long totXD = (long)kRows * kD;
  const int posmod = kNQ * kD;
  const int qt = (kNQ + 15) / 16;  // 19 tiles of 16

  ref_out_kernel<<<(kRows + 255) / 256, 256, 0, stream>>>(refp, out + kRefOff);

  const float* curx = tgt;
  for (int l = 0; l < kNL; ++l) {
    const float* sa_w_in  = p_sa_w_in  + (size_t)l * kQKV * kD;
    const float* sa_b_in  = p_sa_b_in  + (size_t)l * kQKV;
    const float* sa_w_out = p_sa_w_out + (size_t)l * kD * kD;
    const float* sa_b_out = p_sa_b_out + (size_t)l * kD;
    const float* off_w = p_off_w + (size_t)l * 128 * kD;
    const float* off_b = p_off_b + (size_t)l * 128;
    const float* aw_w  = p_aw_w  + (size_t)l * 128 * kD;
    const float* aw_b  = p_aw_b  + (size_t)l * 128;
    const float* val_w = p_val_w + (size_t)l * kD * kD;
    const float* val_b = p_val_b + (size_t)l * kD;
    const float* mso_w = p_mso_w + (size_t)l * kD * kD;
    const float* mso_b = p_mso_b + (size_t)l * kD;
    const float* ff1_w = p_ff1_w + (size_t)l * kDFF * kD;
    const float* ff1_b = p_ff1_b + (size_t)l * kDFF;
    const float* ff2_w = p_ff2_w + (size_t)l * kD * kDFF;
    const float* ff2_b = p_ff2_b + (size_t)l * kD;
    const float* n1_w = p_n1_w + (size_t)l * kD;
    const float* n1_b = p_n1_b + (size_t)l * kD;
    const float* n2_w = p_n2_w + (size_t)l * kD;
    const float* n2_b = p_n2_b + (size_t)l * kD;
    const float* n3_w = p_n3_w + (size_t)l * kD;
    const float* n3_b = p_n3_b + (size_t)l * kD;

    // q = x + query_pos ; Q,K projections use q, V projection uses x
    add_pos_kernel<<<(totXD + 255) / 256, 256, 0, stream>>>(curx, qpos, bq, totXD, posmod);
    gemm(bq, sa_w_in, sa_b_in, bqkv, kRows, 2 * kD, kD, kQKV, 0, stream);
    gemm(curx, sa_w_in + (size_t)2 * kD * kD, sa_b_in + 2 * kD, bqkv + 2 * kD,
         kRows, kD, kD, kQKV, 0, stream);

    attn_scores_kernel<<<dim3(qt, qt, kBS * kH), 32, 0, stream>>>(bqkv, battn);
    softmax_mask_kernel<<<(kBS * kH * kNQ) / 8, 256, 0, stream>>>(battn, q_mask);
    attn_apply_kernel<<<dim3((kNQ + 127) / 128, kBS * kH), 256, 0, stream>>>(battn, bqkv, bt2);
    gemm(bt2, sa_w_out, sa_b_out, bq, kRows, kD, kD, kD, 0, stream);  // t2 -> bq

    // x = LN(output + t2)    (n2)
    ln_res_kernel<<<kRows / 8, 256, 0, stream>>>(curx, bq, n2_w, n2_b, bxA, kRows);

    // --- ms-deform attention ---
    add_pos_kernel<<<(totXD + 255) / 256, 256, 0, stream>>>(bxA, qpos, bq, totXD, posmod);
    gemm(src, val_w, val_b, bval, kBS * kS, kD, kD, kD, 0, stream);
    mask_value_kernel<<<kBS * kS, kD, 0, stream>>>(bval, src_pad);
    gemm(bq, off_w, off_b, boff, kRows, kH * kLV * kP, kD, kH * kLV * kP, 0, stream);
    gemm(bq, aw_w, aw_b, baw, kRows, kH * kLV * kP, kD, kH * kLV * kP, 0, stream);
    msdeform_kernel<<<(kBS * kNQ * kH) / 8, 256, 0, stream>>>(
        boff, baw, bval, refp, svr, tshape, bms, out + kLocOff, out + kAwOff, l);
    gemm(bms, mso_w, mso_b, bt2, kRows, kD, kD, kD, 0, stream);

    // x = LN(x + t2)    (n1)
    ln_res_kernel<<<kRows / 8, 256, 0, stream>>>(bxA, bt2, n1_w, n1_b, bq, kRows);

    // FFN + n3
    gemm(bq, ff1_w, ff1_b, bff, kRows, kDFF, kD, kDFF, 1, stream);
    gemm(bff, ff2_w, ff2_b, bt2, kRows, kD, kDFF, kD, 0, stream);
    float* xout = (l & 1) ? bx1 : bx0;
    ln_res_kernel<<<kRows / 8, 256, 0, stream>>>(bq, bt2, n3_w, n3_b, xout, kRows);

    copy_kernel<<<((totXD / 4) + 255) / 256, 256, 0, stream>>>(
        xout, out + (long)l * totXD, totXD / 4);
    curx = xout;
  }
}
